// Futurecaster_67147518706111
// MI455X (gfx1250) — matrix operand order repaired and checked by reading, not compiled, not run on hardware
//
#include <hip/hip_runtime.h>

// ---------------------------------------------------------------------------
// Types for CDNA5 WMMA (wave32): v_wmma_f32_16x16x32_bf16
// ---------------------------------------------------------------------------
typedef __attribute__((ext_vector_type(16))) __bf16 v16bf;
typedef __attribute__((ext_vector_type(8)))  __bf16 v8bf;
typedef __attribute__((ext_vector_type(8)))  float  v8f;
typedef unsigned short ushort_t;

#define B_SZ   2048
#define T_IN   64
#define F_IN   16
#define T_OUT  32
#define H_SZ   512
#define R_SZ   256
#define N4H    (4 * H_SZ)          // 2048
#define XSTRIDE (T_IN * F_IN)      // 1024 floats per batch row
#define ROWU   40                  // ushorts per padded LDS row (80 B)

__device__ __forceinline__ ushort_t f2bf_rne(float f) {
  union { float f; unsigned int u; } v; v.f = f;
  unsigned int r = v.u + 0x7FFFu + ((v.u >> 16) & 1u);
  return (ushort_t)(r >> 16);
}

__device__ __forceinline__ float sigmoidf_(float x) {
  return 1.0f / (1.0f + __expf(-x));
}

__device__ __forceinline__ void wmma_bf16(v16bf a, v16bf b, v8f& acc) {
  acc = __builtin_amdgcn_wmma_f32_16x16x32_bf16(
      false, a, false, b, (short)0, acc, false, false);
}

// A fragment (contiguous K order): lane l holds row m0+(l&15);
// l<16: K {k0..k0+15}; l>=16: K {k0+16..k0+31}.
__device__ __forceinline__ v16bf load_afrag(const ushort_t* p, int k0) {
  v8bf lo = *(const v8bf*)(p + k0);
  v8bf hi = *(const v8bf*)(p + k0 + 8);
  return __builtin_shufflevector(lo, hi,
      0, 1, 2, 3, 4, 5, 6, 7, 8, 9, 10, 11, 12, 13, 14, 15);
}

// CDNA5 async global -> LDS copy (ASYNCcnt-tracked), 16B per lane.
// VDST = per-lane LDS byte address, VADDR = per-lane global address.
__device__ __forceinline__ void async_b128(unsigned ldsoff, const ushort_t* gp) {
  asm volatile("global_load_async_to_lds_b128 %0, %1, off"
               :: "v"(ldsoff), "v"(gp) : "memory");
}
__device__ __forceinline__ void wait_async0() {
  asm volatile("s_wait_asynccnt 0x0" ::: "memory");
}

// Split workgroup barrier WITHOUT the full fence __syncthreads() implies:
// drain only DScnt (our LDS reads must have executed before others may
// overwrite the buffer); ASYNCcnt is drained separately; global loadcnt
// (A-fragment prefetch) deliberately stays in flight across the barrier.
__device__ __forceinline__ void block_barrier_lds() {
  asm volatile("s_wait_dscnt 0x0\n\t"
               "s_barrier_signal -1\n\t"
               "s_barrier_wait -1" ::: "memory");
}

// ---------------------------------------------------------------------------
// Setup kernels
// ---------------------------------------------------------------------------
__global__ void cvt_bf16_kernel(const float* __restrict__ src,
                                ushort_t* __restrict__ dst, int n) {
  int i = blockIdx.x * blockDim.x + threadIdx.x;
  if (i < n) dst[i] = f2bf_rne(src[i]);
}

__global__ void pad_wih_kernel(const float* __restrict__ src,
                               ushort_t* __restrict__ dst) {
  int i = blockIdx.x * blockDim.x + threadIdx.x;   // N4H*32 threads
  int row = i >> 5, k = i & 31;
  dst[i] = (k < F_IN) ? f2bf_rne(src[row * F_IN + k]) : (ushort_t)0;
}

__global__ void zero_state_kernel(float* __restrict__ c,
                                  ushort_t* __restrict__ hbf) {
  int i = blockIdx.x * blockDim.x + threadIdx.x;   // B*H threads
  c[i] = 0.0f;
  hbf[i] = 0;
}

__global__ void init_p_kernel(const float* __restrict__ x,
                              float* __restrict__ p) {
  int b = blockIdx.x * blockDim.x + threadIdx.x;   // B threads
  if (b < B_SZ) p[b] = x[b * XSTRIDE + (T_IN - 1) * F_IN + 0];
}

// ---------------------------------------------------------------------------
// Gates GEMM with LDS-staged B (async copies):
//   Out[2048, N4H] = A_bf16 @ Whh_bf16^T (+ x @ WihP^T when HAS_X)
// Block = 8 waves = 256(M) x 64(N). Per K-chunk the 64x32 bf16 B strip (4 KB)
// is staged ONCE per block via global_load_async_to_lds_b128 (double-buffered,
// rows padded to 80 B), then each wave reads fragments with ds_load_b128.
// A fragments stay register-double-buffered from global and ride across the
// per-chunk barrier (no loadcnt drain).
// ---------------------------------------------------------------------------
template<bool HAS_X>
__global__ __launch_bounds__(256, 1)
void gemm_lds_kernel(const ushort_t* __restrict__ Abf,   // [2048,512] bf16
                     const ushort_t* __restrict__ Wbf,   // [N4H,512] bf16
                     float* __restrict__ Out,            // [2048,N4H]
                     const float* __restrict__ X,        // x base (HAS_X)
                     const ushort_t* __restrict__ WihP,  // [N4H,32] bf16
                     int xoff) {
  __shared__ ushort_t lbuf[2][64 * ROWU];                // 2 x 5 KB

  const int tid  = threadIdx.x;
  const int lane = tid & 31;
  const int wave = tid >> 5;
  const int m0   = (blockIdx.y * 8 + wave) * 32;
  const int n0   = blockIdx.x * 64;
  const int lm   = lane & 15;
  const int hi   = lane >> 4;

  // async producer: thread moves one 16B segment per chunk
  const int prow = tid >> 2;                             // 0..63
  const int pseg = tid & 3;                              // 0..3
  const ushort_t* gsrc = Wbf + (n0 + prow) * H_SZ + pseg * 8;
  const unsigned ldst0 = (unsigned)(uintptr_t)&lbuf[0][prow * ROWU + pseg * 8];
  const unsigned ldst1 = (unsigned)(uintptr_t)&lbuf[1][prow * ROWU + pseg * 8];

  const ushort_t* arow0 = Abf + (m0 + lm) * H_SZ + hi * 16;
  const ushort_t* arow1 = arow0 + 16 * H_SZ;

  v8f acc[8] = {};

  if (HAS_X) {
    // seed accumulators with x @ WihP^T (one K=32 WMMA per tile pair)
    const float* xr0 = X + (m0 + lm) * XSTRIDE + xoff;
    const float* xr1 = xr0 + 16 * XSTRIDE;
    v16bf ax0, ax1;
    #pragma unroll
    for (int i = 0; i < 8; i++) {
      ax0[i] = (__bf16)xr0[i];  ax0[8 + i] = (__bf16)xr0[8 + i];
      ax1[i] = (__bf16)xr1[i];  ax1[8 + i] = (__bf16)xr1[8 + i];
    }
    v16bf xb0 = *(const v16bf*)(WihP + (n0 + 0 * 16 + lm) * 32 + hi * 16);
    v16bf xb1 = *(const v16bf*)(WihP + (n0 + 1 * 16 + lm) * 32 + hi * 16);
    v16bf xb2 = *(const v16bf*)(WihP + (n0 + 2 * 16 + lm) * 32 + hi * 16);
    v16bf xb3 = *(const v16bf*)(WihP + (n0 + 3 * 16 + lm) * 32 + hi * 16);
    wmma_bf16(ax0, xb0, acc[0]); wmma_bf16(ax0, xb1, acc[1]);
    wmma_bf16(ax0, xb2, acc[2]); wmma_bf16(ax0, xb3, acc[3]);
    wmma_bf16(ax1, xb0, acc[4]); wmma_bf16(ax1, xb1, acc[5]);
    wmma_bf16(ax1, xb2, acc[6]); wmma_bf16(ax1, xb3, acc[7]);
  }

  // stage chunk 0
  async_b128(ldst0, gsrc);
  v16bf a0 = load_afrag(arow0, 0);
  v16bf a1 = load_afrag(arow1, 0);
  wait_async0();
  block_barrier_lds();

  #pragma unroll
  for (int ch = 0; ch < 16; ++ch) {
    const int k0 = ch * 32;
    if (ch < 15)                                   // prefetch next B strip
      async_b128((ch & 1) ? ldst0 : ldst1, gsrc + k0 + 32);

    const ushort_t* lb = &lbuf[ch & 1][0];
    v16bf b0 = *(const v16bf*)(lb + (0 * 16 + lm) * ROWU + hi * 16);
    v16bf b1 = *(const v16bf*)(lb + (1 * 16 + lm) * ROWU + hi * 16);
    v16bf b2 = *(const v16bf*)(lb + (2 * 16 + lm) * ROWU + hi * 16);
    v16bf b3 = *(const v16bf*)(lb + (3 * 16 + lm) * ROWU + hi * 16);

    v16bf na0 = a0, na1 = a1;
    if (ch < 15) {                                 // prefetch next A frags
      na0 = load_afrag(arow0, k0 + 32);
      na1 = load_afrag(arow1, k0 + 32);
    }

    wmma_bf16(a0, b0, acc[0]); wmma_bf16(a0, b1, acc[1]);
    wmma_bf16(a0, b2, acc[2]); wmma_bf16(a0, b3, acc[3]);
    wmma_bf16(a1, b0, acc[4]); wmma_bf16(a1, b1, acc[5]);
    wmma_bf16(a1, b2, acc[6]); wmma_bf16(a1, b3, acc[7]);

    a0 = na0; a1 = na1;
    if (ch < 15) {
      wait_async0();        // next buffer fully written (per-wave ASYNCcnt)
      block_barrier_lds();  // dscnt-only drain + split barrier
    }
  }

  // Epilogue: D layout — VGPR r, lane l: M = mbase + 8*hi + r, N = n0+16t+lm
  #pragma unroll
  for (int mt = 0; mt < 2; mt++) {
    int mbase = m0 + mt * 16 + hi * 8;
    #pragma unroll
    for (int t = 0; t < 4; t++) {
      int n = n0 + t * 16 + lm;
      #pragma unroll
      for (int r = 0; r < 8; r++)
        Out[(mbase + r) * N4H + n] = acc[mt * 4 + t][r];
    }
  }
}

// ---------------------------------------------------------------------------
// Register-path WMMA GEMM (used for the regressor: N=256, bias+ReLU fused)
// ---------------------------------------------------------------------------
template<bool RELU>
__global__ __launch_bounds__(256, 1)
void gemm_wmma_kernel(const ushort_t* __restrict__ Abf,   // [2048,512] bf16
                      const ushort_t* __restrict__ Wbf,   // [N,512] bf16
                      const float*    __restrict__ bias,  // [N]
                      float*          __restrict__ Out,   // [2048,N]
                      int N) {
  const int lane = threadIdx.x & 31;
  const int wave = threadIdx.x >> 5;
  const int m0   = (blockIdx.y * 8 + wave) * 32;
  const int n0   = blockIdx.x * 64;
  const int lm   = lane & 15;
  const int hi   = lane >> 4;

  const ushort_t* arow0 = Abf + (m0 + lm) * H_SZ + hi * 16;
  const ushort_t* arow1 = arow0 + 16 * H_SZ;
  const ushort_t* br0 = Wbf + (n0 + 0 * 16 + lm) * H_SZ + hi * 16;
  const ushort_t* br1 = Wbf + (n0 + 1 * 16 + lm) * H_SZ + hi * 16;
  const ushort_t* br2 = Wbf + (n0 + 2 * 16 + lm) * H_SZ + hi * 16;
  const ushort_t* br3 = Wbf + (n0 + 3 * 16 + lm) * H_SZ + hi * 16;

  v8f acc[8] = {};

  v16bf a0 = load_afrag(arow0, 0);
  v16bf a1 = load_afrag(arow1, 0);
  v16bf b0 = *(const v16bf*)(br0);
  v16bf b1 = *(const v16bf*)(br1);
  v16bf b2 = *(const v16bf*)(br2);
  v16bf b3 = *(const v16bf*)(br3);

  #pragma unroll
  for (int k0 = 32; k0 < H_SZ; k0 += 32) {
    v16bf na0 = load_afrag(arow0, k0);
    v16bf na1 = load_afrag(arow1, k0);
    v16bf nb0 = *(const v16bf*)(br0 + k0);
    v16bf nb1 = *(const v16bf*)(br1 + k0);
    v16bf nb2 = *(const v16bf*)(br2 + k0);
    v16bf nb3 = *(const v16bf*)(br3 + k0);

    wmma_bf16(a0, b0, acc[0]); wmma_bf16(a0, b1, acc[1]);
    wmma_bf16(a0, b2, acc[2]); wmma_bf16(a0, b3, acc[3]);
    wmma_bf16(a1, b0, acc[4]); wmma_bf16(a1, b1, acc[5]);
    wmma_bf16(a1, b2, acc[6]); wmma_bf16(a1, b3, acc[7]);

    a0 = na0; a1 = na1; b0 = nb0; b1 = nb1; b2 = nb2; b3 = nb3;
  }
  wmma_bf16(a0, b0, acc[0]); wmma_bf16(a0, b1, acc[1]);
  wmma_bf16(a0, b2, acc[2]); wmma_bf16(a0, b3, acc[3]);
  wmma_bf16(a1, b0, acc[4]); wmma_bf16(a1, b1, acc[5]);
  wmma_bf16(a1, b2, acc[6]); wmma_bf16(a1, b3, acc[7]);

  #pragma unroll
  for (int mt = 0; mt < 2; mt++) {
    int mbase = m0 + mt * 16 + hi * 8;
    #pragma unroll
    for (int t = 0; t < 4; t++) {
      int n = n0 + t * 16 + lm;
      float bv = bias[n];
      #pragma unroll
      for (int r = 0; r < 8; r++) {
        float v = acc[mt * 4 + t][r] + bv;
        if (RELU) v = fmaxf(v, 0.0f);
        Out[(mbase + r) * N + n] = v;
      }
    }
  }
}

// ---------------------------------------------------------------------------
// LSTM pointwise kernels
// ---------------------------------------------------------------------------
__global__ void enc_act_kernel(const float* __restrict__ gates,
                               const float* __restrict__ bvec,  // enc_b [N4H]
                               float* __restrict__ c,
                               ushort_t* __restrict__ hbf) {
  int id = blockIdx.x * blockDim.x + threadIdx.x;   // B*H
  int b = id >> 9, u = id & (H_SZ - 1);
  const float* g = gates + b * N4H;
  float gi = g[u]            + bvec[u];
  float gf = g[H_SZ + u]     + bvec[H_SZ + u];
  float gg = g[2 * H_SZ + u] + bvec[2 * H_SZ + u];
  float go = g[3 * H_SZ + u] + bvec[3 * H_SZ + u];
  float cv = sigmoidf_(gf) * c[id] + sigmoidf_(gi) * tanhf(gg);
  c[id] = cv;
  hbf[id] = f2bf_rne(sigmoidf_(go) * tanhf(cv));
}

__global__ void dec_act_kernel(const float* __restrict__ gates,
                               const float* __restrict__ dWih,  // [N4H]
                               const float* __restrict__ dB,    // [N4H]
                               const float* __restrict__ p,     // [B]
                               float* __restrict__ c,
                               ushort_t* __restrict__ hbf) {
  int id = blockIdx.x * blockDim.x + threadIdx.x;
  int b = id >> 9, u = id & (H_SZ - 1);
  const float* g = gates + b * N4H;
  float pb = p[b];
  float gi = g[u]            + pb * dWih[u]            + dB[u];
  float gf = g[H_SZ + u]     + pb * dWih[H_SZ + u]     + dB[H_SZ + u];
  float gg = g[2 * H_SZ + u] + pb * dWih[2 * H_SZ + u] + dB[2 * H_SZ + u];
  float go = g[3 * H_SZ + u] + pb * dWih[3 * H_SZ + u] + dB[3 * H_SZ + u];
  float cv = sigmoidf_(gf) * c[id] + sigmoidf_(gi) * tanhf(gg);
  c[id] = cv;
  hbf[id] = f2bf_rne(sigmoidf_(go) * tanhf(cv));
}

// ---------------------------------------------------------------------------
// Final dot: p[b] = relu_r[b,:] . W2 + b2 ; one wave per batch row (wave32)
// ---------------------------------------------------------------------------
__global__ __launch_bounds__(256)
void dot_out_kernel(const float* __restrict__ rbuf,   // [B, R]
                    const float* __restrict__ W2,     // [R]
                    const float* __restrict__ b2,     // [1]
                    float* __restrict__ p,            // [B]
                    float* __restrict__ out,          // [B, T_OUT]
                    int t) {
  int lane = threadIdx.x & 31;
  int wave = threadIdx.x >> 5;
  int b = blockIdx.x * 8 + wave;
  const float* r = rbuf + b * R_SZ;
  float s = 0.0f;
  #pragma unroll
  for (int i = 0; i < R_SZ / 32; i++)
    s += r[lane + i * 32] * W2[lane + i * 32];
  #pragma unroll
  for (int off = 16; off > 0; off >>= 1)
    s += __shfl_xor(s, off, 32);
  if (lane == 0) {
    float v = s + b2[0];
    p[b] = v;
    out[b * T_OUT + t] = v;
  }
}

// ---------------------------------------------------------------------------
// Host-side orchestration (graph-capture safe: stream launches only)
// ---------------------------------------------------------------------------
extern "C" void kernel_launch(void* const* d_in, const int* in_sizes, int n_in,
                              void* d_out, int out_size, void* d_ws, size_t ws_size,
                              hipStream_t stream) {
  (void)in_sizes; (void)n_in; (void)out_size; (void)ws_size;
  const float* x       = (const float*)d_in[0];
  const float* enc_Wih = (const float*)d_in[1];
  const float* enc_Whh = (const float*)d_in[2];
  const float* enc_b   = (const float*)d_in[3];
  const float* dec_Wih = (const float*)d_in[4];
  const float* dec_Whh = (const float*)d_in[5];
  const float* dec_b   = (const float*)d_in[6];
  const float* W1      = (const float*)d_in[7];
  const float* b1      = (const float*)d_in[8];
  const float* W2      = (const float*)d_in[9];
  const float* b2      = (const float*)d_in[10];
  float* out = (float*)d_out;

  // Workspace carve-up (all offsets 256B aligned)
  char* ws = (char*)d_ws;
  size_t off = 0;
  auto alloc = [&](size_t bytes) {
    void* p = ws + off;
    off += (bytes + 255) & ~(size_t)255;
    return p;
  };
  float*    gates = (float*)   alloc((size_t)B_SZ * N4H * 4);   // 16 MB
  float*    cbuf  = (float*)   alloc((size_t)B_SZ * H_SZ * 4);  // 4 MB
  ushort_t* hbf   = (ushort_t*)alloc((size_t)B_SZ * H_SZ * 2);  // 2 MB
  float*    rbuf  = (float*)   alloc((size_t)B_SZ * R_SZ * 4);  // 2 MB
  float*    pbuf  = (float*)   alloc((size_t)B_SZ * 4);
  ushort_t* WhhEb = (ushort_t*)alloc((size_t)N4H * H_SZ * 2);   // 2 MB
  ushort_t* WhhDb = (ushort_t*)alloc((size_t)N4H * H_SZ * 2);   // 2 MB
  ushort_t* W1b   = (ushort_t*)alloc((size_t)R_SZ * H_SZ * 2);
  ushort_t* WihP  = (ushort_t*)alloc((size_t)N4H * 32 * 2);

  const int TB = 256;
  // --- setup: weight conversion + state/p init (ws is poisoned before timing)
  cvt_bf16_kernel<<<(N4H * H_SZ + TB - 1) / TB, TB, 0, stream>>>(enc_Whh, WhhEb, N4H * H_SZ);
  cvt_bf16_kernel<<<(N4H * H_SZ + TB - 1) / TB, TB, 0, stream>>>(dec_Whh, WhhDb, N4H * H_SZ);
  cvt_bf16_kernel<<<(R_SZ * H_SZ + TB - 1) / TB, TB, 0, stream>>>(W1, W1b, R_SZ * H_SZ);
  pad_wih_kernel<<<(N4H * 32 + TB - 1) / TB, TB, 0, stream>>>(enc_Wih, WihP);
  zero_state_kernel<<<(B_SZ * H_SZ) / TB, TB, 0, stream>>>(cbuf, hbf);
  init_p_kernel<<<(B_SZ + TB - 1) / TB, TB, 0, stream>>>(x, pbuf);

  dim3 gGates(N4H / 64, B_SZ / 256);   // (32, 8): block = 256M x 64N
  dim3 gReg(R_SZ / 64, B_SZ / 256);    // (4, 8)
  const int actBlocks = (B_SZ * H_SZ) / TB;

  // --- encoder: 64 recurrent steps (x term folded in as WMMA seed)
  for (int t = 0; t < T_IN; t++) {
    gemm_lds_kernel<true><<<gGates, TB, 0, stream>>>(
        hbf, WhhEb, gates, x, WihP, t * F_IN);
    enc_act_kernel<<<actBlocks, TB, 0, stream>>>(gates, enc_b, cbuf, hbf);
  }

  // --- decoder: 32 steps; step 0 input = x[:, -1, 0]
  for (int t = 0; t < T_OUT; t++) {
    gemm_lds_kernel<false><<<gGates, TB, 0, stream>>>(
        hbf, WhhDb, gates, nullptr, nullptr, 0);
    dec_act_kernel<<<actBlocks, TB, 0, stream>>>(gates, dec_Wih, dec_b, pbuf, cbuf, hbf);
    gemm_wmma_kernel<true><<<gReg, TB, 0, stream>>>(hbf, W1b, b1, rbuf, R_SZ);
    dot_out_kernel<<<B_SZ / 8, TB, 0, stream>>>(rbuf, W2, b2, pbuf, out, t);
  }
}
